// RelPartialLearnableMultiHeadAttn_53712861004254
// MI455X (gfx1250) — matrix operand order repaired and checked by reading, not compiled, not run on hardware
//
#include <hip/hip_runtime.h>

typedef __attribute__((ext_vector_type(16))) __bf16 v16bf;
typedef __attribute__((ext_vector_type(8)))  __bf16 v8bf;
typedef __attribute__((ext_vector_type(8)))  float  v8f;

#define WMMA_BF16(A, B, C) \
  __builtin_amdgcn_wmma_f32_16x16x32_bf16(false, (A), false, (B), (short)0, (C), false, false)

// ---------------------------------------------------------------------------
// A-fragment loader: 16x32 bf16 A-matrix per ISA 7.12.2.
// lanes 0-15 : M=lane, e0..7 -> K=k..k+7,    e8..15 -> K=k+16..k+23
// lanes16-31 : M=lane-16, K offset +8. Two 16-byte loads.
// koff = k0 + (lane<16 ? 0 : 8)
// ---------------------------------------------------------------------------
__device__ __forceinline__ v16bf afrag_load(const __bf16* rowptr, int koff) {
  v8bf lo = *(const v8bf*)(rowptr + koff);
  v8bf hi = *(const v8bf*)(rowptr + koff + 16);
  v16bf a;
#pragma unroll
  for (int i = 0; i < 8; ++i) { a[i] = lo[i]; a[i + 8] = hi[i]; }
  return a;
}

__device__ __forceinline__ v16bf bfrag_gather(const __bf16* base, size_t stride, int klo) {
  v16bf b;
#pragma unroll
  for (int s = 0; s < 16; ++s) { int kk = (s < 8) ? (klo + s) : (klo + 8 + s); b[s] = base[(size_t)kk * stride]; }
  return b;
}

// ---------------------------------------------------------------------------
// Elementwise conversion / generation kernels
// ---------------------------------------------------------------------------
__global__ void cvt_bf16_kernel(const float* __restrict__ in, __bf16* __restrict__ out, int n) {
  int i = blockIdx.x * blockDim.x + threadIdx.x;
  if (i < n) out[i] = (__bf16)in[i];
}

// out[k*1024+j] = in[j*1024+k]  (W^T in bf16, so GEMM B-frags are contiguous)
__global__ void cvt_tr_bf16_kernel(const float* __restrict__ in, __bf16* __restrict__ out) {
  int idx = blockIdx.x * blockDim.x + threadIdx.x;
  int k = idx >> 10, j = idx & 1023;
  out[idx] = (__bf16)in[j * 1024 + k];
}

// sinusoidal relative position embeddings, [4095][1024] bf16
__global__ void gen_r_kernel(__bf16* __restrict__ r) {
  int idx = blockIdx.x * blockDim.x + threadIdx.x;
  if (idx >= 4095 * 512) return;
  int m = idx / 512, c = idx % 512;
  float pos  = (float)(m - 2047);
  float invf = __expf(-(float)(2 * c) * (9.210340371976184f / 1024.0f)); // 10000^{-2c/D}
  float ang  = pos * invf;
  r[(size_t)m * 1024 + 2 * c]     = (__bf16)__sinf(ang);
  r[(size_t)m * 1024 + 2 * c + 1] = (__bf16)__cosf(ang);
}

// zero the pad slots of Rr: per (h,d) row of width 4128, slots [0,15] and [4111,4127]
__global__ void zero_pad_kernel(__bf16* __restrict__ Rr) {
  int idx = blockIdx.x * blockDim.x + threadIdx.x;
  if (idx >= 16 * 64 * 33) return;
  int row = idx / 33, s = idx % 33;
  int slot = (s < 16) ? s : (4111 + (s - 16));
  Rr[(size_t)row * 4128 + slot] = (__bf16)0.0f;
}

// ---------------------------------------------------------------------------
// Projection GEMM: out = A(bf16, Mx1024 row-major) @ BT(bf16, 1024x1024 K-major)
// One wave computes a 16x64 tile, K loop in steps of 32. Mode epilogues:
//  0: q -> qu/qv [16][2048][64] with u/v bias    1: k -> kT [16][64][2048]
//  2: v -> v    [16][2048][64]                   3: r -> Rr [16*64][4128] reversed
// ---------------------------------------------------------------------------
__global__ __launch_bounds__(128) void gemm_proj_kernel(
    const __bf16* __restrict__ A, const __bf16* __restrict__ BT,
    int Mrows, int Mtiles, int mode,
    const float* __restrict__ ub, const float* __restrict__ vb,
    __bf16* __restrict__ o0, __bf16* __restrict__ o1) {
  int lane = threadIdx.x & 31;
  int wid  = blockIdx.x * (blockDim.x >> 5) + (threadIdx.x >> 5);
  int mt = wid >> 4, nt = wid & 15;
  if (mt >= Mtiles) return;
  int I0 = mt * 16, N0 = nt * 64;
  int Ml = lane & 15;
  int row = I0 + Ml; if (row > Mrows - 1) row = Mrows - 1;
  const __bf16* rowptr = A + (size_t)row * 1024;
  int klo = (lane < 16) ? 0 : 8;

  v8f c[4];
#pragma unroll
  for (int n = 0; n < 4; ++n)
#pragma unroll
    for (int r = 0; r < 8; ++r) c[n][r] = 0.0f;

  for (int k0 = 0; k0 < 1024; k0 += 32) {
    v16bf a = afrag_load(rowptr, k0 + klo);
#pragma unroll
    for (int n = 0; n < 4; ++n) {
      v16bf b = bfrag_gather(BT + (size_t)k0 * 1024 + N0 + n * 16 + Ml, 1024, klo);
      c[n] = WMMA_BF16(a, b, c[n]);
    }
  }

  int g = lane >> 4, N = lane & 15;
#pragma unroll
  for (int n = 0; n < 4; ++n) {
#pragma unroll
    for (int r = 0; r < 8; ++r) {
      int i = I0 + r + 8 * g;
      int j = N0 + n * 16 + N;
      float val = c[n][r];
      if (mode == 0) {
        size_t idx = (((size_t)(j >> 6)) * 2048 + i) * 64 + (j & 63);
        o0[idx] = (__bf16)(val + ub[j]);
        o1[idx] = (__bf16)(val + vb[j]);
      } else if (mode == 1) {
        o0[(size_t)j * 2048 + i] = (__bf16)val;
      } else if (mode == 2) {
        o0[(((size_t)(j >> 6)) * 2048 + i) * 64 + (j & 63)] = (__bf16)val;
      } else {
        if (i < 4095) {
          int t = 4094 - i;                 // Rr[h][d][16+t] = rp[h][4094-t][d]
          o0[(size_t)j * 4128 + 16 + t] = (__bf16)val;
        }
      }
    }
  }
}

// ---------------------------------------------------------------------------
// Fused rel-attention: one wave per (head, 16-row query block).
// Online softmax over all 2048 keys in chunks of 32.
// rel-shift semantics (exact reference trace):
//   j <= i   : BD = (q_i+vb) . rp[2Lq-2-(i-j)]   -> Rr slot 16 + (i-j... t=i-j)
//   j == i+1 : BD = 0
//   j >  i+1 : BD = (q_{i+1}+vb) . rp[j-i-2]     -> t = 2Lq-(j-i)
// ---------------------------------------------------------------------------
__global__ __launch_bounds__(32) void attn_kernel(
    const __bf16* __restrict__ qu, const __bf16* __restrict__ qv,
    const __bf16* __restrict__ kT, const __bf16* __restrict__ vv,
    const __bf16* __restrict__ Rr, float* __restrict__ out) {
  __shared__ float  sBD[16 * 32];
  __shared__ float  sBDu[16 * 32];
  __shared__ __bf16 sP[16 * 32];

  int lane = threadIdx.x & 31;
  int h = blockIdx.x >> 7;
  int I = (blockIdx.x & 127) << 4;
  int Ml = lane & 15, g = lane >> 4, N = Ml;
  int klo = (lane < 16) ? 0 : 8;
  const float scale = 0.125f;

  const __bf16* qurow = qu + ((size_t)h * 2048 + I + Ml) * 64;
  const __bf16* qvrow = qv + ((size_t)h * 2048 + I + Ml) * 64;
  int rowu = I + 1 + Ml; if (rowu > 2047) rowu = 2047;   // shifted rows for leak band
  const __bf16* qvrowU = qv + ((size_t)h * 2048 + rowu) * 64;

  v16bf au0 = afrag_load(qurow, klo),       au1 = afrag_load(qurow, 32 + klo);
  v16bf av0 = afrag_load(qvrow, klo),       av1 = afrag_load(qvrow, 32 + klo);
  v16bf aw0 = afrag_load(qvrowU, klo),      aw1 = afrag_load(qvrowU, 32 + klo);

  const __bf16* kTh = kT + (size_t)h * 64 * 2048;
  const __bf16* Rrh = Rr + (size_t)h * 64 * 4128;
  const __bf16* vh  = vv + (size_t)h * 2048 * 64;

  v8f acc[4];
  float mrun[8], lrun[8];
#pragma unroll
  for (int n = 0; n < 4; ++n)
#pragma unroll
    for (int r = 0; r < 8; ++r) acc[n][r] = 0.0f;
#pragma unroll
  for (int r = 0; r < 8; ++r) { mrun[r] = -3.0e38f; lrun[r] = 0.0f; }

  for (int Jc = 0; Jc < 2048; Jc += 32) {
    float s2[2][8];
#pragma unroll
    for (int hh = 0; hh < 2; ++hh) {
      int J = Jc + hh * 16;
      // ---- AC tile ----
      v16bf b0 = bfrag_gather(kTh + J + Ml, 2048, klo);
      v16bf b1 = bfrag_gather(kTh + (size_t)32 * 2048 + J + Ml, 2048, klo);
      v8f cc;
#pragma unroll
      for (int r = 0; r < 8; ++r) cc[r] = 0.0f;
      cc = WMMA_BF16(au0, b0, cc);
      cc = WMMA_BF16(au1, b1, cc);

      // ---- BD bands ----
      bool doL = (J <= I), doU = (J >= I);
      if (doL) {
        int t0 = I - J;                         // band slots t0 .. t0+31 => t in [t0-16, t0+15]
        v16bf r00 = bfrag_gather(Rrh + t0 + Ml, 4128, klo);
        v16bf r01 = bfrag_gather(Rrh + (size_t)32 * 4128 + t0 + Ml, 4128, klo);
        v16bf r10 = bfrag_gather(Rrh + 16 + t0 + Ml, 4128, klo);
        v16bf r11 = bfrag_gather(Rrh + (size_t)32 * 4128 + 16 + t0 + Ml, 4128, klo);
        v8f tA, tB;
#pragma unroll
        for (int r = 0; r < 8; ++r) { tA[r] = 0.0f; tB[r] = 0.0f; }
        tA = WMMA_BF16(av0, r00, tA); tA = WMMA_BF16(av1, r01, tA);
        tB = WMMA_BF16(av0, r10, tB); tB = WMMA_BF16(av1, r11, tB);
#pragma unroll
        for (int r = 0; r < 8; ++r) {
          int M = r + 8 * g;
          sBD[M * 32 + N]      = tA[r];
          sBD[M * 32 + 16 + N] = tB[r];
        }
      }
      if (doU) {
        int t0u = 4096 - (J - I);               // upper (leak) band
        v16bf u00 = bfrag_gather(Rrh + t0u + Ml, 4128, klo);
        v16bf u01 = bfrag_gather(Rrh + (size_t)32 * 4128 + t0u + Ml, 4128, klo);
        v16bf u10 = bfrag_gather(Rrh + 16 + t0u + Ml, 4128, klo);
        v16bf u11 = bfrag_gather(Rrh + (size_t)32 * 4128 + 16 + t0u + Ml, 4128, klo);
        v8f tA, tB;
#pragma unroll
        for (int r = 0; r < 8; ++r) { tA[r] = 0.0f; tB[r] = 0.0f; }
        tA = WMMA_BF16(aw0, u00, tA); tA = WMMA_BF16(aw1, u01, tA);
        tB = WMMA_BF16(aw0, u10, tB); tB = WMMA_BF16(aw1, u11, tB);
#pragma unroll
        for (int r = 0; r < 8; ++r) {
          int M = r + 8 * g;
          sBDu[M * 32 + N]      = tA[r];
          sBDu[M * 32 + 16 + N] = tB[r];
        }
      }
      asm volatile("s_wait_dscnt 0" ::: "memory");

#pragma unroll
      for (int r = 0; r < 8; ++r) {
        int M = r + 8 * g;
        int delta = (J - I) + N - M;            // j - i
        int c = M - N + 16;                     // band column, in [1,31]
        float bdv = 0.0f;
        if (delta <= 0)      bdv = sBD[M * 32 + c];
        else if (delta >= 2) bdv = sBDu[M * 32 + c];
        s2[hh][r] = (cc[r] + bdv) * scale;
      }
      asm volatile("" ::: "memory");
    }

    // ---- online softmax over the 32-column chunk ----
    float p0[8], p1[8];
#pragma unroll
    for (int r = 0; r < 8; ++r) {
      float v = fmaxf(s2[0][r], s2[1][r]);
      v = fmaxf(v, __shfl_xor(v, 1, 32));
      v = fmaxf(v, __shfl_xor(v, 2, 32));
      v = fmaxf(v, __shfl_xor(v, 4, 32));
      v = fmaxf(v, __shfl_xor(v, 8, 32));
      float mnew = fmaxf(mrun[r], v);
      float corr = __expf(mrun[r] - mnew);
      mrun[r] = mnew;
      p0[r] = __expf(s2[0][r] - mnew);
      p1[r] = __expf(s2[1][r] - mnew);
      float rs = p0[r] + p1[r];
      rs += __shfl_xor(rs, 1, 32);
      rs += __shfl_xor(rs, 2, 32);
      rs += __shfl_xor(rs, 4, 32);
      rs += __shfl_xor(rs, 8, 32);
      lrun[r] = lrun[r] * corr + rs;
#pragma unroll
      for (int n = 0; n < 4; ++n) acc[n][r] *= corr;
    }

    // ---- P to LDS (C-layout -> A-layout transpose) ----
#pragma unroll
    for (int r = 0; r < 8; ++r) {
      int M = r + 8 * g;
      sP[M * 32 + N]      = (__bf16)p0[r];
      sP[M * 32 + 16 + N] = (__bf16)p1[r];
    }
    asm volatile("s_wait_dscnt 0" ::: "memory");
    v16bf pf = afrag_load((const __bf16*)&sP[(size_t)Ml * 32], klo);

    // ---- P @ V ----
#pragma unroll
    for (int n = 0; n < 4; ++n) {
      v16bf vbf = bfrag_gather(vh + (size_t)Jc * 64 + n * 16 + Ml, 64, klo);
      acc[n] = WMMA_BF16(pf, vbf, acc[n]);
    }
    asm volatile("" ::: "memory");
  }

  // ---- epilogue: out[i, h*64 + d] ----
#pragma unroll
  for (int n = 0; n < 4; ++n) {
#pragma unroll
    for (int r = 0; r < 8; ++r) {
      int i = I + r + 8 * g;
      out[(size_t)i * 1024 + h * 64 + n * 16 + N] = acc[n][r] / lrun[r];
    }
  }
}

// ---------------------------------------------------------------------------
// Host-side orchestration
// ---------------------------------------------------------------------------
extern "C" void kernel_launch(void* const* d_in, const int* in_sizes, int n_in,
                              void* d_out, int out_size, void* d_ws, size_t ws_size,
                              hipStream_t stream) {
  const float* w  = (const float*)d_in[0];
  const float* Wq = (const float*)d_in[1];
  const float* Wk = (const float*)d_in[2];
  const float* Wv = (const float*)d_in[3];
  const float* Wr = (const float*)d_in[4];
  const float* ub = (const float*)d_in[5];
  const float* vb = (const float*)d_in[6];
  float* out = (float*)d_out;

  char* ws = (char*)d_ws;
  size_t off = 0;
  __bf16* w_bf  = (__bf16*)(ws + off); off += (size_t)2048 * 1024 * 2;
  __bf16* WqT   = (__bf16*)(ws + off); off += (size_t)1024 * 1024 * 2;
  __bf16* WkT   = (__bf16*)(ws + off); off += (size_t)1024 * 1024 * 2;
  __bf16* WvT   = (__bf16*)(ws + off); off += (size_t)1024 * 1024 * 2;
  __bf16* WrT   = (__bf16*)(ws + off); off += (size_t)1024 * 1024 * 2;
  __bf16* r_bf  = (__bf16*)(ws + off); off += (size_t)4095 * 1024 * 2;
  off = (off + 255) & ~(size_t)255;
  __bf16* qu_bf = (__bf16*)(ws + off); off += (size_t)16 * 2048 * 64 * 2;
  __bf16* qv_bf = (__bf16*)(ws + off); off += (size_t)16 * 2048 * 64 * 2;
  __bf16* kT_bf = (__bf16*)(ws + off); off += (size_t)16 * 64 * 2048 * 2;
  __bf16* v_bf  = (__bf16*)(ws + off); off += (size_t)16 * 2048 * 64 * 2;
  __bf16* Rr_bf = (__bf16*)(ws + off); off += (size_t)16 * 64 * 4128 * 2;

  // 1. conversions / generation
  cvt_bf16_kernel<<<8192, 256, 0, stream>>>(w, w_bf, 2048 * 1024);
  cvt_tr_bf16_kernel<<<4096, 256, 0, stream>>>(Wq, WqT);
  cvt_tr_bf16_kernel<<<4096, 256, 0, stream>>>(Wk, WkT);
  cvt_tr_bf16_kernel<<<4096, 256, 0, stream>>>(Wv, WvT);
  cvt_tr_bf16_kernel<<<4096, 256, 0, stream>>>(Wr, WrT);
  gen_r_kernel<<<8190, 256, 0, stream>>>(r_bf);
  zero_pad_kernel<<<132, 256, 0, stream>>>(Rr_bf);

  // 2. projections (WMMA GEMMs)
  gemm_proj_kernel<<<512, 128, 0, stream>>>(w_bf, WqT, 2048, 128, 0, ub, vb, qu_bf, qv_bf);
  gemm_proj_kernel<<<512, 128, 0, stream>>>(w_bf, WkT, 2048, 128, 1, ub, vb, kT_bf, kT_bf);
  gemm_proj_kernel<<<512, 128, 0, stream>>>(w_bf, WvT, 2048, 128, 2, ub, vb, v_bf, v_bf);
  gemm_proj_kernel<<<1024, 128, 0, stream>>>(r_bf, WrT, 4095, 256, 3, ub, vb, Rr_bf, Rr_bf);

  // 3. fused relative attention (16 heads x 128 row blocks)
  attn_kernel<<<2048, 32, 0, stream>>>(qu_bf, qv_bf, kT_bf, v_bf, Rr_bf, out);
}
